// MimiVectorQuantization_54322746359967
// MI455X (gfx1250) — matrix operand order repaired and checked by reading, not compiled, not run on hardware
//
#include <hip/hip_runtime.h>
#include <hip/hip_bf16.h>
#include <stdint.h>

// ---------------- problem constants ----------------
#define IN_DIM   512
#define CB_D     256
#define CB_K     2048
#define T_ROWS   131072

typedef __attribute__((ext_vector_type(16))) __bf16 v16bf;
typedef __attribute__((ext_vector_type(8)))  float  v8f;
typedef __attribute__((ext_vector_type(2)))  float  v2f;
typedef __attribute__((ext_vector_type(2)))  __bf16 v2bf;

union BFrag { v16bf v; uint4 q[2]; };
union AFrag { v16bf v; unsigned u[8]; };

__device__ __forceinline__ v8f wmma_bf16(v16bf a, v16bf b, v8f c) {
    // D = A(16x32) * B(32x16) + C, f32 accum
    return __builtin_amdgcn_wmma_f32_16x16x32_bf16(false, a, false, b, (short)0, c, false, false);
}

// A fragment (16x32 bf16) from a row-major f32 matrix with row stride `stride`,
// tile origin (row0, k0).  ISA 16-bit A layout: lane m=l&15 holds row m;
// VGPR r holds K pair k0 + (r<4 ? 2r : 2r+8) + 8*(l>>4).
// Uses native packed f32->bf16 conversion (convertvector) instead of bit math.
__device__ __forceinline__ v16bf load_a_f32(const float* __restrict__ base, int stride,
                                            int row0, int k0, int lane) {
    const int m = lane & 15, h = lane >> 4;
    const float* row = base + (size_t)(row0 + m) * stride + k0;
    AFrag a;
#pragma unroll
    for (int r = 0; r < 8; ++r) {
        const int kk = (r < 4 ? 2 * r : 2 * r + 8) + 8 * h;
        v2f f = *(const v2f*)(row + kk);
        a.u[r] = __builtin_bit_cast(unsigned, __builtin_convertvector(f, v2bf));
    }
    return a.v;
}

// A fragment from a row-major bf16 matrix (16 rows x `stride` cols), tile at col k0.
__device__ __forceinline__ v16bf load_a_bf16(const __bf16* __restrict__ base, int stride,
                                             int k0, int lane) {
    const int m = lane & 15, h = lane >> 4;
    const __bf16* row = base + (size_t)m * stride + k0;
    AFrag a;
#pragma unroll
    for (int r = 0; r < 8; ++r) {
        const int kk = (r < 4 ? 2 * r : 2 * r + 8) + 8 * h;
        a.u[r] = *(const unsigned*)(row + kk);
    }
    return a.v;
}

// B fragment (32x16 bf16) from a row-major bf16 matrix; `tile` points at (k_row0, n_col0),
// row stride `stride`.  Lane l holds N column (l&15); element s holds K = (s<8 ? s : s+8) + 8*(l>>4).
__device__ __forceinline__ v16bf load_b_bf16(const __bf16* __restrict__ tile, int stride, int lane) {
    const int n = lane & 15, h = lane >> 4;
    BFrag b;
#pragma unroll
    for (int s = 0; s < 16; ++s) {
        const int k = (s < 8 ? s : s + 8) + 8 * h;
        b.v[s] = tile[(size_t)k * stride + n];
    }
    return b.v;
}

// ---------------------------------------------------------------------------
// Tensor Data Mover: DMA a [256 rows x 16 cols] bf16 tile of emb_t (row stride
// 2048) from global memory into LDS at byte offset `lds_off` (row-major 256x16).
// Descriptor per CDNA5 ISA 8.3/8.4: 2D tensor, data_size=2B, type=2.
// Toolchain probe: 6-arg builtin form (g0, g1, g2, g3, g4, cpol).
// ---------------------------------------------------------------------------
#if defined(__has_builtin)
#if __has_builtin(__builtin_amdgcn_tensor_load_to_lds) && __has_builtin(__builtin_amdgcn_s_wait_tensorcnt)
#define HAVE_TDM 1
#endif
#endif

#ifdef HAVE_TDM
typedef __attribute__((ext_vector_type(4))) unsigned int tdm_u32x4;
typedef __attribute__((ext_vector_type(8))) int          tdm_i32x8;
typedef __attribute__((ext_vector_type(4))) int          tdm_i32x4;

__device__ __forceinline__ void tdm_load_emb_tile(const __bf16* gsrc, unsigned lds_off) {
    const unsigned long long ga = (unsigned long long)(uintptr_t)gsrc;
    tdm_u32x4 g0;
    g0.x = 1u;                                  // count=1 (valid user descriptor)
    g0.y = lds_off;                             // lds_addr (bytes)
    g0.z = (unsigned)ga;                        // global_addr[31:0]
    g0.w = (unsigned)(ga >> 32) | (2u << 30);   // global_addr[56:32] | type=2 (image)
    tdm_i32x8 g1;
    g1[0] = 1 << 16;             // workgroup_mask=0 | data_size=1 (2 bytes)
    g1[1] = (int)(2048u << 16);  // atomic_barrier_addr=0 | tensor_dim0 lo16 = 2048
    g1[2] = (int)(256u << 16);   // tensor_dim0 hi16=0 | tensor_dim1 lo16 = 256
    g1[3] = (int)(16u << 16);    // tensor_dim1 hi16=0 | tile_dim0 = 16
    g1[4] = 256;                 // tile_dim1 = 256 | tile_dim2 = 0
    g1[5] = 2048;                // tensor_dim0_stride lo32 = 2048
    g1[6] = 0;                   // stride0 hi16 = 0 | tensor_dim1_stride lo16 = 0
    g1[7] = 0;                   // tensor_dim1_stride hi32 = 0
    const tdm_i32x4 z4 = {0, 0, 0, 0};                      // D# groups 2/3: unused (2D tensor)
    const tdm_i32x8 z8 = {0, 0, 0, 0, 0, 0, 0, 0};
    __builtin_amdgcn_tensor_load_to_lds(g0, g1, z4, z4, z8, 0);
}
#endif

// ---------------------------------------------------------------------------
// prep_convert: build bf16 transposed weights + e_half = 0.5*||e_k||^2
//   w_in_t  [IN_DIM][CB_D]  = w_in^T   (bf16)
//   w_out_t [CB_D][IN_DIM]  = w_out^T  (bf16)
//   emb_t   [CB_D][CB_K]    = emb^T    (bf16)
// ---------------------------------------------------------------------------
__global__ void prep_convert(const float* __restrict__ w_in, const float* __restrict__ w_out,
                             const float* __restrict__ emb,
                             __bf16* __restrict__ w_in_t, __bf16* __restrict__ w_out_t,
                             __bf16* __restrict__ emb_t, float* __restrict__ e_half) {
    const int N1 = IN_DIM * CB_D;      // w_in_t
    const int N2 = CB_D * IN_DIM;      // w_out_t
    const int N3 = CB_D * CB_K;        // emb_t
    const int N4 = CB_K;               // e_half
    const int total = N1 + N2 + N3 + N4;
    for (int idx = blockIdx.x * blockDim.x + threadIdx.x; idx < total;
         idx += gridDim.x * blockDim.x) {
        int j = idx;
        if (j < N1) {               // [i][d] <- w_in[d][i]
            int i = j / CB_D, d = j % CB_D;
            w_in_t[j] = (__bf16)w_in[(size_t)d * IN_DIM + i];
            continue;
        }
        j -= N1;
        if (j < N2) {               // [d][i] <- w_out[i][d]
            int d = j / IN_DIM, i = j % IN_DIM;
            w_out_t[j] = (__bf16)w_out[(size_t)i * CB_D + d];
            continue;
        }
        j -= N2;
        if (j < N3) {               // [d][k] <- emb[k][d]
            int d = j / CB_K, k = j % CB_K;
            emb_t[j] = (__bf16)emb[(size_t)k * CB_D + d];
            continue;
        }
        j -= N3;
        {
            const float* e = emb + (size_t)j * CB_D;
            float s = 0.f;
            for (int d = 0; d < CB_D; ++d) s += e[d] * e[d];
            e_half[j] = 0.5f * s;
        }
    }
}

// ---------------------------------------------------------------------------
// prep_gemm: code_out[k][i] = sum_d emb[k][d] * w_out[i][d]   (2048x512, WMMA bf16)
// One wave per 16x16 output tile; contraction D=256 = 8 WMMA steps.
// ---------------------------------------------------------------------------
__global__ __launch_bounds__(128) void prep_gemm(const float* __restrict__ emb,
                                                 const __bf16* __restrict__ w_out_t,
                                                 float* __restrict__ code_out) {
    const int g    = blockIdx.x * 4 + (threadIdx.x >> 5);   // global wave-tile id
    const int lane = threadIdx.x & 31;
    const int c0 = (g >> 5) * 16;        // code tile   (2048/16 = 128)
    const int i0 = (g & 31) * 16;        // out-dim tile (512/16 = 32)
    const int m = lane & 15, h = lane >> 4;

    v8f c = {};
#pragma unroll
    for (int k2 = 0; k2 < 8; ++k2) {
        v16bf a = load_a_f32(emb, CB_D, c0, k2 * 32, lane);
        v16bf b = load_b_bf16(w_out_t + (size_t)(k2 * 32) * IN_DIM + i0, IN_DIM, lane);
        c = wmma_bf16(a, b, c);
    }
#pragma unroll
    for (int r = 0; r < 8; ++r)
        code_out[(size_t)(c0 + r + 8 * h) * IN_DIM + i0 + m] = c[r];
}

// ---------------------------------------------------------------------------
// vq_fused: per block (4 wave32s, 64 T-rows):
//   stage 1: xp(64x256) = x @ w_in^T via WMMA, w_in k-slices staged in LDS
//   stage 2: score all 2048 codes via WMMA; emb tiles DMA'd into LDS with the
//            Tensor Data Mover, double-buffered (prefetch ct+1 while scoring ct);
//            per-lane running argmax of (x.e - 0.5||e||^2), shfl-reduce at end
//   stage 3: out rows = code_out[idx] gather (float4, coalesced)
// ---------------------------------------------------------------------------
__global__ __launch_bounds__(128) void vq_fused(const float* __restrict__ x,
                                                const __bf16* __restrict__ w_in_t,
                                                const __bf16* __restrict__ emb_t,
                                                const float* __restrict__ e_half,
                                                const float* __restrict__ code_out,
                                                float* __restrict__ out,
                                                int* __restrict__ indices) {
    // LDS layout (48.25 KB):
    //   [0, 32K)        xp staging: 4 waves x 16 rows x 256 cols bf16
    //   [32K, 48K)      stage-1 w_in slice (32x256 bf16) | stage-2 emb tiles 2x8K (TDM dbl-buf)
    //   [48K, 48K+256)  per-wave chosen indices
    __shared__ __align__(16) char smem[32768 + 16384 + 256];
    __bf16* lds_xp  = (__bf16*)smem;
    __bf16* lds_B1  = (__bf16*)(smem + 32768);
    int*    lds_idx = (int*)(smem + 32768 + 16384);

    const int tid  = threadIdx.x;
    const int wave = tid >> 5;
    const int lane = tid & 31;
    const int m = lane & 15, h = lane >> 4;
    const int t0 = (blockIdx.x * 4 + wave) * 16;   // this wave's 16 T-rows

    // ---------------- stage 1: xp = x @ w_in^T ----------------
    v8f acc[16];
    const v8f vzero = {};
#pragma unroll
    for (int dt = 0; dt < 16; ++dt) acc[dt] = vzero;

    for (int kt = 0; kt < 16; ++kt) {               // K = 512 in 16 slices of 32
        __syncthreads();
        // cooperative stage of w_in_t rows [kt*32, kt*32+32) x 256 cols (16 KB)
        {
            const uint4* src = (const uint4*)(w_in_t + (size_t)kt * 32 * CB_D);
            uint4* dst = (uint4*)lds_B1;
#pragma unroll
            for (int i = 0; i < 8; ++i) dst[tid + 128 * i] = src[tid + 128 * i];
        }
        __syncthreads();
        v16bf a = load_a_f32(x, IN_DIM, t0, kt * 32, lane);
#pragma unroll
        for (int dt = 0; dt < 16; ++dt) {
            v16bf b = load_b_bf16(lds_B1 + dt * 16, CB_D, lane);
            acc[dt] = wmma_bf16(a, b, acc[dt]);
        }
    }

    // write xp tile to this wave's LDS region as bf16 (C/D layout: VGPR r -> row r+8h, col = lane&15)
    __bf16* myxp = lds_xp + (size_t)wave * 16 * CB_D;
#pragma unroll
    for (int dt = 0; dt < 16; ++dt) {
#pragma unroll
        for (int r = 0; r < 8; ++r)
            myxp[(size_t)(r + 8 * h) * CB_D + dt * 16 + m] = (__bf16)acc[dt][r];
    }

    // preload xp as A fragments (contraction D = 256 -> 8 fragments, resident)
    v16bf a2[8];
#pragma unroll
    for (int k2 = 0; k2 < 8; ++k2) a2[k2] = load_a_bf16(myxp, CB_D, k2 * 32, lane);

    float best_v[8];
    int   best_i[8];
#pragma unroll
    for (int r = 0; r < 8; ++r) { best_v[r] = -3.4e38f; best_i[r] = 0; }

    // ---------------- stage 2: score 2048 codes, track argmax ----------------
#ifdef HAVE_TDM
    __syncthreads();                                // all waves done reading lds_B1 (stage 1)
    if (wave == 0)                                  // prefetch first tile into buffer 0
        tdm_load_emb_tile(emb_t, (unsigned)(uintptr_t)(smem + 32768));
#pragma unroll 1
    for (int ct = 0; ct < CB_K / 16; ++ct) {
        __syncthreads();                            // consumers done with buffer T(ct+1) targets
        if (wave == 0) {
            if (ct + 1 < CB_K / 16) {
                tdm_load_emb_tile(emb_t + (ct + 1) * 16,
                                  (unsigned)(uintptr_t)(smem + 32768 + ((ct + 1) & 1) * 8192));
                __builtin_amdgcn_s_wait_tensorcnt(1);   // T(ct) landed (in-order)
            } else {
                __builtin_amdgcn_s_wait_tensorcnt(0);
            }
        }
        __syncthreads();                            // T(ct) visible to all waves
        const __bf16* B2 = (const __bf16*)(smem + 32768 + (ct & 1) * 8192);
#else
#pragma unroll 1
    for (int ct = 0; ct < CB_K / 16; ++ct) {
        __bf16* B2w = (__bf16*)(smem + 32768);
        __syncthreads();                            // previous B2 consumers done
#pragma unroll
        for (int i = 0; i < 4; ++i) {               // cooperative 8 KB stage
            const int q = tid + 128 * i;
            const int row = q >> 1, hs = q & 1;
            *(uint4*)(B2w + row * 16 + hs * 8) =
                *(const uint4*)(emb_t + (size_t)row * CB_K + ct * 16 + hs * 8);
        }
        __syncthreads();
        const __bf16* B2 = B2w;
#endif
        v8f c = vzero;
#pragma unroll
        for (int k2 = 0; k2 < 8; ++k2) {
            v16bf b = load_b_bf16(B2 + k2 * 32 * 16, 16, lane);
            c = wmma_bf16(a2[k2], b, c);
        }
        const float eh = e_half[ct * 16 + m];
        const int   ci = ct * 16 + m;
#pragma unroll
        for (int r = 0; r < 8; ++r) {
            const float v = c[r] - eh;              // x.e - 0.5||e||^2
            if (v > best_v[r]) { best_v[r] = v; best_i[r] = ci; }
        }
    }

    // reduce argmax across the 16 lanes holding one row (xor <= 8 stays in the group)
#pragma unroll
    for (int r = 0; r < 8; ++r) {
#pragma unroll
        for (int off = 1; off < 16; off <<= 1) {
            const float ov = __shfl_xor(best_v[r], off, 32);
            const int   oi = __shfl_xor(best_i[r], off, 32);
            if (ov > best_v[r] || (ov == best_v[r] && oi < best_i[r])) {
                best_v[r] = ov; best_i[r] = oi;
            }
        }
    }

    if (m == 0) {                                   // lanes 0 and 16: rows r and r+8
#pragma unroll
        for (int r = 0; r < 8; ++r) {
            const int row = r + 8 * h;
            lds_idx[wave * 16 + row] = best_i[r];
            indices[t0 + row] = best_i[r];
        }
    }
    // same-wave LDS RAW below is kept in order by hardware dscnt waits

    // ---------------- stage 3: out rows = code_out[idx] gather ----------------
#pragma unroll 1
    for (int row = 0; row < 16; ++row) {
        const int idx = lds_idx[wave * 16 + row];
        const float4* src = (const float4*)(code_out + (size_t)idx * IN_DIM);
        float4* dst = (float4*)(out + (size_t)(t0 + row) * IN_DIM);
#pragma unroll
        for (int j = 0; j < 4; ++j) dst[lane + 32 * j] = src[lane + 32 * j];
    }
}

// ---------------------------------------------------------------------------
extern "C" void kernel_launch(void* const* d_in, const int* in_sizes, int n_in,
                              void* d_out, int out_size, void* d_ws, size_t ws_size,
                              hipStream_t stream) {
    const float* x     = (const float*)d_in[0];   // (T, 512)
    const float* w_in  = (const float*)d_in[1];   // (256, 512)
    const float* w_out = (const float*)d_in[2];   // (512, 256)
    const float* emb   = (const float*)d_in[3];   // (2048, 256)

    float* out     = (float*)d_out;                            // T*512 f32
    int*   indices = (int*)(out + (size_t)T_ROWS * IN_DIM);    // T int32 (concatenated)

    // workspace carve-up (≈5.6 MB total)
    char* ws = (char*)d_ws;
    __bf16* w_in_t  = (__bf16*)(ws);                       // 512*256*2 = 256 KB
    __bf16* w_out_t = (__bf16*)(ws + 262144);              // 256*512*2 = 256 KB
    __bf16* emb_t   = (__bf16*)(ws + 524288);              // 256*2048*2 = 1 MB
    float*  e_half  = (float*)(ws + 1572864);              // 2048*4 = 8 KB
    float*  code_out = (float*)(ws + 1581056);             // 2048*512*4 = 4 MB

    {   // transposed bf16 weights + half code norms
        const int total = IN_DIM * CB_D * 2 + CB_D * CB_K + CB_K;
        prep_convert<<<(total + 255) / 256, 256, 0, stream>>>(w_in, w_out, emb, w_in_t,
                                                              w_out_t, emb_t, e_half);
    }
    // code_out = emb @ w_out^T : 128 x 32 tiles of 16x16, 4 waves/block
    prep_gemm<<<(128 * 32) / 4, 128, 0, stream>>>(emb, w_out_t, code_out);

    // fused project + score + argmin + gather: 64 rows/block
    vq_fused<<<T_ROWS / 64, 128, 0, stream>>>(x, w_in_t, emb_t, e_half, code_out, out, indices);
}
